// MotionLSTM_5394478924393
// MI455X (gfx1250) — matrix operand order repaired and checked by reading, not compiled, not run on hardware
//
#include <hip/hip_runtime.h>
#include <hip/hip_bf16.h>

typedef __attribute__((ext_vector_type(16))) _Float16 v16h;
typedef __attribute__((ext_vector_type(8)))  _Float16 v8h;
typedef __attribute__((ext_vector_type(8)))  float    v8f;
typedef __attribute__((ext_vector_type(4)))  unsigned int u32x4;
typedef __attribute__((ext_vector_type(8)))  unsigned int u32x8;

#define Bsz 256
#define Tn  512
#define Dn  64
#define Hn  128
#define Gn  512   // 4*H
#define Cn  16
#define BT  16    // batch rows per workgroup
#define NTHREADS 512

__device__ __forceinline__ float sigmoidf_(float x) { return 1.0f / (1.0f + __expf(-x)); }

__device__ __forceinline__ unsigned rfl_(unsigned v) {
  return (unsigned)__builtin_amdgcn_readfirstlane((int)v);
}

// ---------------------------------------------------------------------------
// Tensor Data Mover: async 2-D tile load Global -> LDS (CDNA5 §8).
// Group0: count=1 | lds_addr | global_addr[56:0] | type=2.
// Group1: data_size | tensor_dim0/1 = tile_dim0/1 (tile == viewed tensor) |
//         tensor_dim0_stride = row stride. 2-group form (<=2D), tracked by
//         TENSORcnt; wait with s_wait_tensorcnt.
// data_size_code: 1 = 2 bytes (fp16), 2 = 4 bytes (fp32).
// ---------------------------------------------------------------------------
__device__ __forceinline__ void tdm_load_2d(unsigned lds_byte_off, const void* gptr,
                                            unsigned tile_w_elems, unsigned tile_h_rows,
                                            unsigned row_stride_elems, unsigned data_size_code) {
  unsigned long long ga = (unsigned long long)(uintptr_t)gptr;
  u32x4 g0;
  g0[0] = rfl_(1u);                                             // count=1 (user D#)
  g0[1] = rfl_(lds_byte_off);                                   // lds_addr (bytes)
  g0[2] = rfl_((unsigned)ga);                                   // global_addr[31:0]
  g0[3] = rfl_(((unsigned)(ga >> 32) & 0x01FFFFFFu) | (2u << 30)); // addr[56:32] | type=2
  u32x8 g1;
  g1[0] = rfl_(data_size_code << 16);                           // wg_mask=0 | data_size
  g1[1] = rfl_(tile_w_elems << 16);                             // [63:48] = tensor_dim0 lo
  g1[2] = rfl_(tile_h_rows << 16);                              // [79:64]=dim0 hi(0) | [95:80]=dim1 lo
  g1[3] = rfl_(tile_w_elems << 16);                             // [111:96]=dim1 hi(0) | [127:112]=tile_dim0
  g1[4] = rfl_(tile_h_rows);                                    // [143:128]=tile_dim1 | tile_dim2=0
  g1[5] = rfl_(row_stride_elems);                               // tensor_dim0_stride[31:0]
  g1[6] = rfl_(0u);                                             // stride hi | dim1_stride lo
  g1[7] = rfl_(0u);
  asm volatile("tensor_load_to_lds %0, %1" :: "s"(g0), "s"(g1) : "memory");
}

// A-matrix 16x32 fp16 chunk from row-major fp16 LDS (ISA 16-bit A layout).
__device__ __forceinline__ v16h load_A16x32(const _Float16* __restrict__ base, int stride, int k0) {
  const int lane = threadIdx.x & 31;
  const int row  = lane & 15;
  const int hi   = (lane >> 4) & 1;
  const _Float16* p = base + row * stride + k0 + hi * 8;
  v8h lo = *(const v8h*)(p);
  v8h hh = *(const v8h*)(p + 16);
  v16h a;
  #pragma unroll
  for (int e = 0; e < 8; ++e) { a[e] = lo[e]; a[e + 8] = hh[e]; }
  return a;
}

// Same A chunk but sourced from a raw fp32 staged tile (layer 0 x data),
// converting at load time: 4x ds_load_b128 + cvt, no extra LDS pass.
__device__ __forceinline__ v16h load_A16x32_f32(const float* __restrict__ base, int stride, int k0) {
  const int lane = threadIdx.x & 31;
  const int row  = lane & 15;
  const int hi   = (lane >> 4) & 1;
  const float* p = base + row * stride + k0 + hi * 8;
  v8f lo = *(const v8f*)(p);
  v8f hh = *(const v8f*)(p + 16);
  v16h a;
  #pragma unroll
  for (int e = 0; e < 8; ++e) { a[e] = (_Float16)lo[e]; a[e + 8] = (_Float16)hh[e]; }
  return a;
}

// B-matrix 32x16 chunk (K x N) loaded ONCE from global fp32 weights [G][stride]
// into registers: B[k][n] = w[n0+n][k0+k]; per lane two runs of 8 floats (k = 8h+s, 16+8h+s).
__device__ __forceinline__ v16h load_B32x16_global(const float* __restrict__ w, int stride,
                                                   int n0, int k0) {
  const int lane = threadIdx.x & 31;
  const int n    = lane & 15;
  const int hi   = (lane >> 4) & 1;
  const float* p = w + (size_t)(n0 + n) * stride + k0 + hi * 8;
  v16h b;
  #pragma unroll
  for (int e = 0; e < 16; ++e) b[e] = (_Float16)p[e < 8 ? e : e + 8];
  return b;
}

// ---------------------------------------------------------------------------
// Fused LSTM layer over the full T=512 recurrence.
//  - weights: fp16, register-resident per wave (12-16 v16h)
//  - input tile: TDM double-buffered in LDS (async, TENSORcnt)
//  - h: fp16 in LDS; c: fp32 in registers (4 cells/thread)
//  - gates: v_wmma_f32_16x16x32_f16, 2 gate N-tiles per wave
// ---------------------------------------------------------------------------
template<int DIN, int LAYER, int FINAL>
__global__ __launch_bounds__(NTHREADS)
void lstm_layer_kernel(const float*    __restrict__ x,     // [B,T,D]   (LAYER==0)
                       const _Float16* __restrict__ hin,   // [T,B,H]   (LAYER==1)
                       const float*    __restrict__ w_ih,  // [G,DIN]
                       const float*    __restrict__ w_hh,  // [G,H]
                       const float*    __restrict__ b_ih,  // [G]
                       const float*    __restrict__ b_hh,  // [G]
                       _Float16*       __restrict__ hout,  // [T,B,H]   (LAYER==0)
                       const float*    __restrict__ fc_w,  // [C,H]     (FINAL)
                       const float*    __restrict__ fc_b,  // [C]       (FINAL)
                       float*          __restrict__ out)   // [B,C]     (FINAL)
{
  constexpr int NKI = DIN / 32;          // input-side K chunks
  constexpr int NKH = Hn  / 32;          // hidden-side K chunks
  constexpr int STAGE_BYTES = BT * DIN * ((LAYER == 0) ? 4 : 2);

  extern __shared__ char smem_raw[];
  // NOTE: no LDS-pointer arrays (addrspacecast in an aggregate initializer
  // breaks device linking); derive the stage pointer from parity on demand.
  _Float16* hld    = (_Float16*)(smem_raw + 2 * STAGE_BYTES); // [BT][H]
  float*    gates  = (float*)(hld + BT * Hn);                 // [BT][G]
  float*    bias   = gates + BT * Gn;                         // [G]

  const int tid  = threadIdx.x;
  const int wave = tid >> 5;
  const int lane = tid & 31;
  const int b0   = blockIdx.x * BT;

  // Kick off the t=0 tile DMA as early as possible.
  if (wave == 0) {
    if (LAYER == 0)
      tdm_load_2d((unsigned)(uintptr_t)smem_raw, x + (size_t)b0 * Tn * Dn,
                  Dn, BT, (unsigned)(Tn * Dn), 2u);
    else
      tdm_load_2d((unsigned)(uintptr_t)smem_raw, hin + (size_t)b0 * Hn,
                  Hn, BT, (unsigned)Hn, 1u);
  }

  // Hoist this wave's weight tiles into registers (one-time global reads).
  v16h wreg[2][NKI + NKH];
  #pragma unroll
  for (int tt = 0; tt < 2; ++tt) {
    const int n0 = (wave + tt * 16) * 16;
    #pragma unroll
    for (int k = 0; k < NKI; ++k) wreg[tt][k]       = load_B32x16_global(w_ih, DIN, n0, k * 32);
    #pragma unroll
    for (int k = 0; k < NKH; ++k) wreg[tt][NKI + k] = load_B32x16_global(w_hh, Hn,  n0, k * 32);
  }

  for (int i = tid; i < Gn;      i += NTHREADS) bias[i] = b_ih[i] + b_hh[i];
  for (int i = tid; i < BT * Hn; i += NTHREADS) hld[i]  = (_Float16)0.0f;

  float cst[4] = {0.f, 0.f, 0.f, 0.f};  // persistent cell state: 4 cells/thread

  for (int t = 0; t < Tn; ++t) {
    // Current tile DMA must be complete (no-op wait for waves != 0)...
    __builtin_amdgcn_s_wait_tensorcnt(0);
    // ...then prefetch next step's tile into the other buffer. Safe: every wave
    // already passed last step's mid-barrier, so reads of that buffer are done.
    if (wave == 0 && t + 1 < Tn) {
      unsigned nxt_off = (unsigned)(uintptr_t)smem_raw + (unsigned)(((t + 1) & 1) * STAGE_BYTES);
      if (LAYER == 0)
        tdm_load_2d(nxt_off, x + ((size_t)b0 * Tn + (t + 1)) * Dn,
                    Dn, BT, (unsigned)(Tn * Dn), 2u);
      else
        tdm_load_2d(nxt_off, hin + ((size_t)(t + 1) * Bsz + b0) * Hn,
                    Hn, BT, (unsigned)Hn, 1u);
    }
    __syncthreads();  // tile t visible; orders prev step's hld writes too

    const char* cur = smem_raw + (t & 1) * STAGE_BYTES;

    // --- gate GEMM: register-resident B, LDS A, 2 gate N-tiles per wave ---
    #pragma unroll
    for (int tt = 0; tt < 2; ++tt) {
      const int n0 = (wave + tt * 16) * 16;
      v8f acc = {};
      #pragma unroll
      for (int k = 0; k < NKI; ++k) {
        v16h a = (LAYER == 0) ? load_A16x32_f32((const float*)cur, DIN, k * 32)
                              : load_A16x32((const _Float16*)cur, DIN, k * 32);
        acc = __builtin_amdgcn_wmma_f32_16x16x32_f16(false, a, false, wreg[tt][k],
                                                     (short)0, acc, false, false);
      }
      #pragma unroll
      for (int k = 0; k < NKH; ++k) {
        v16h a = load_A16x32(hld, Hn, k * 32);
        acc = __builtin_amdgcn_wmma_f32_16x16x32_f16(false, a, false, wreg[tt][NKI + k],
                                                     (short)0, acc, false, false);
      }
      const int n = lane & 15, hi = (lane >> 4) & 1;
      #pragma unroll
      for (int r = 0; r < 8; ++r) gates[(size_t)(r + hi * 8) * Gn + n0 + n] = acc[r];
    }
    __syncthreads();

    // --- LSTM cell elementwise update: 4 fixed cells per thread ---
    #pragma unroll
    for (int k = 0; k < 4; ++k) {
      const int cell = tid + k * NTHREADS;   // 0..2047
      const int row  = cell >> 7;
      const int col  = cell & 127;
      float gi = sigmoidf_(gates[row * Gn +       col] + bias[      col]);
      float gf = sigmoidf_(gates[row * Gn + 128 + col] + bias[128 + col]);
      float gg = tanhf    (gates[row * Gn + 256 + col] + bias[256 + col]);
      float go = sigmoidf_(gates[row * Gn + 384 + col] + bias[384 + col]);
      cst[k] = gf * cst[k] + gi * gg;
      float h = go * tanhf(cst[k]);
      hld[row * Hn + col] = (_Float16)h;
      if (LAYER == 0) hout[((size_t)t * Bsz + b0 + row) * Hn + col] = (_Float16)h;
    }
    // (no barrier here: next iteration's post-wait barrier orders hld/gates reuse)
  }

  if (FINAL) {
    __syncthreads();
    if (wave == 0) {  // FC head: one 16x16x128 WMMA chain per workgroup
      v8f acc = {};
      #pragma unroll
      for (int k = 0; k < NKH; ++k) {
        v16h a = load_A16x32(hld, Hn, k * 32);
        v16h b = load_B32x16_global(fc_w, Hn, 0, k * 32);
        acc = __builtin_amdgcn_wmma_f32_16x16x32_f16(false, a, false, b,
                                                     (short)0, acc, false, false);
      }
      const int n = lane & 15, hi = (lane >> 4) & 1;
      #pragma unroll
      for (int r = 0; r < 8; ++r) {
        int m = r + hi * 8;
        out[(size_t)(b0 + m) * Cn + n] = acc[r] + fc_b[n];
      }
    }
  }
}

extern "C" void kernel_launch(void* const* d_in, const int* in_sizes, int n_in,
                              void* d_out, int out_size, void* d_ws, size_t ws_size,
                              hipStream_t stream) {
  (void)in_sizes; (void)n_in; (void)out_size; (void)ws_size;
  const float* x     = (const float*)d_in[0];
  const float* w_ih0 = (const float*)d_in[1];
  const float* w_hh0 = (const float*)d_in[2];
  const float* b_ih0 = (const float*)d_in[3];
  const float* b_hh0 = (const float*)d_in[4];
  const float* w_ih1 = (const float*)d_in[5];
  const float* w_hh1 = (const float*)d_in[6];
  const float* b_ih1 = (const float*)d_in[7];
  const float* b_hh1 = (const float*)d_in[8];
  const float* fc_w  = (const float*)d_in[9];
  const float* fc_b  = (const float*)d_in[10];

  _Float16* h1 = (_Float16*)d_ws;  // [T][B][H] fp16 intermediate, 32 MB

  // LDS: input double buffer + h tile + gate buffer + bias (~46 KB)
  auto smem_bytes = [](int din, int layer) -> size_t {
    size_t stage = (size_t)BT * din * ((layer == 0) ? 4 : 2);
    return 2 * stage + (size_t)BT * Hn * 2 + (size_t)BT * Gn * 4 + (size_t)Gn * 4;
  };

  lstm_layer_kernel<Dn, 0, 0><<<Bsz / BT, NTHREADS, smem_bytes(Dn, 0), stream>>>(
      x, nullptr, w_ih0, w_hh0, b_ih0, b_hh0, h1, nullptr, nullptr, nullptr);

  lstm_layer_kernel<Hn, 1, 1><<<Bsz / BT, NTHREADS, smem_bytes(Hn, 1), stream>>>(
      nullptr, h1, w_ih1, w_hh1, b_ih1, b_hh1, nullptr, fc_w, fc_b, (float*)d_out);
}
